// EEGformer_5703716569222
// MI455X (gfx1250) — compile-verified, matrix operand order checked by reading, not run on hardware
//
#include <hip/hip_runtime.h>
#include <hip/hip_fp16.h>
#include <math.h>

typedef _Float16 half_t;
typedef __attribute__((ext_vector_type(16))) _Float16 v16h;
typedef __attribute__((ext_vector_type(8)))  float    v8f;

#define C_EEG 127
#define NCF   120
#define BQ    4
#define MDIM  480
#define JDIM  121
#define HEADS 8
#define DH    60
#define NTOK  (BQ * C_EEG * JDIM)   // 61468

// ---------------------------------------------------------------------------
// WMMA helpers (CDNA5 wave32, 16x16x32 f16 -> f32)
// ---------------------------------------------------------------------------
__device__ __forceinline__ v8f wmma_f32(v16h a, v16h b, v8f c) {
  // 8 args: (neg_a, A, neg_b, B, c_mod, C, reuse_a, reuse_b)
  return __builtin_amdgcn_wmma_f32_16x16x32_f16(false, a, false, b, (short)0, c,
                                                false, false);
}

// A-fragment (and B-from-weight-rows fragment), row-major source with leading
// dim `ld`.  ISA 16-bit A layout: lanes 0-15 -> K {0..7,16..23},
// lanes 16-31 -> K {8..15,24..31}, row = lane&15.
__device__ __forceinline__ v16h ld_frag_row(const half_t* p, int ld, int r0, int k0) {
  int lane = threadIdx.x & 31;
  const half_t* q = p + (size_t)(r0 + (lane & 15)) * ld + k0 + ((lane & 16) ? 8 : 0);
  v16h a;
#pragma unroll
  for (int e = 0; e < 8; ++e) a[e] = q[e];
#pragma unroll
  for (int e = 0; e < 8; ++e) a[8 + e] = q[16 + e];
  return a;
}

// B-fragment where the effective B[k,n] is stored row-major over k (ld = row
// stride): column gather per lane.
__device__ __forceinline__ v16h ld_frag_col(const half_t* p, int ld, int n0, int k0) {
  int lane = threadIdx.x & 31;
  int n  = n0 + (lane & 15);
  int kb = k0 + ((lane & 16) ? 8 : 0);
  v16h a;
#pragma unroll
  for (int e = 0; e < 8; ++e) a[e] = p[(size_t)(kb + e) * ld + n];
#pragma unroll
  for (int e = 0; e < 8; ++e) a[8 + e] = p[(size_t)(kb + 16 + e) * ld + n];
  return a;
}

// ---------------------------------------------------------------------------
// Generic WMMA GEMM:  C[M,N] = A[M,K] (f16) * W[N,K]^T (f16), f32 accumulate.
// MODE 0: store f16 to outH               (QKV projection, no bias)
// MODE 1: RTM scatter: ss[...] = val + rtm_bias[...]   (blockIdx.z = batch)
// MODE 2: val+=bias; exact GELU; store f16 (MLP fc1)
// MODE 3: ss[row*480+col] += val + bias    (MLP fc2 + residual)
// ---------------------------------------------------------------------------
template <int MODE>
__global__ __launch_bounds__(256) void gemm_wmma_kernel(
    const half_t* __restrict__ Abase, long long sAb, int lda,
    const half_t* __restrict__ Wbase, long long sWb, int ldw,
    int Mrows, int Ncols, int K,
    float* __restrict__ outF, half_t* __restrict__ outH, int ldo,
    const float* __restrict__ bias, const float* __restrict__ extra)
{
  __shared__ half_t As[128 * 40];   // 32-wide K tile, padded to 40 halves/row
  __shared__ half_t Bs[128 * 40];

  const half_t* A = Abase + (size_t)blockIdx.z * sAb;
  const half_t* W = Wbase + (size_t)blockIdx.z * sWb;
  const int tileM = blockIdx.x * 128;
  const int tileN = blockIdx.y * 128;
  const int wid = threadIdx.x >> 5;        // 8 waves
  const int wm  = wid >> 1;                // 4 row-waves  (32 rows each)
  const int wn  = wid & 1;                 // 2 col-waves  (64 cols each)
  const int lr  = threadIdx.x >> 1;        // load row 0..127
  const int lk  = (threadIdx.x & 1) * 16;  // load k-offset 0/16

  v8f acc[2][4] = {};

  for (int k0 = 0; k0 < K; k0 += 32) {
    {
      int gr = tileM + lr;
      half_t* dst = As + lr * 40 + lk;
      if (gr < Mrows) {
        const half_t* src = A + (size_t)gr * lda + k0 + lk;
#pragma unroll
        for (int e = 0; e < 16; ++e) dst[e] = src[e];
      } else {
#pragma unroll
        for (int e = 0; e < 16; ++e) dst[e] = (half_t)0.f;
      }
    }
    {
      int gn = tileN + lr;
      half_t* dst = Bs + lr * 40 + lk;
      if (gn < Ncols) {
        const half_t* src = W + (size_t)gn * ldw + k0 + lk;
#pragma unroll
        for (int e = 0; e < 16; ++e) dst[e] = src[e];
      } else {
#pragma unroll
        for (int e = 0; e < 16; ++e) dst[e] = (half_t)0.f;
      }
    }
    __syncthreads();

    v16h af[2], bf[4];
#pragma unroll
    for (int mi = 0; mi < 2; ++mi) af[mi] = ld_frag_row(As, 40, 32 * wm + 16 * mi, 0);
#pragma unroll
    for (int ni = 0; ni < 4; ++ni) bf[ni] = ld_frag_row(Bs, 40, 64 * wn + 16 * ni, 0);
#pragma unroll
    for (int mi = 0; mi < 2; ++mi)
#pragma unroll
      for (int ni = 0; ni < 4; ++ni)
        acc[mi][ni] = wmma_f32(af[mi], bf[ni], acc[mi][ni]);
    __syncthreads();
  }

  // epilogue (C/D layout: vgpr e: lanes 0-15 -> M=e, lanes 16-31 -> M=8+e)
  const int lane = threadIdx.x & 31;
  const int rsel = (lane >> 4) * 8;
  const int cn   = lane & 15;
#pragma unroll
  for (int mi = 0; mi < 2; ++mi) {
#pragma unroll
    for (int ni = 0; ni < 4; ++ni) {
      int col = tileN + 64 * wn + 16 * ni + cn;
      if (col >= Ncols) continue;
#pragma unroll
      for (int e = 0; e < 8; ++e) {
        int row = tileM + 32 * wm + 16 * mi + rsel + e;
        if (row >= Mrows) continue;
        float val = acc[mi][ni][e];
        if (MODE == 0) {
          outH[(size_t)row * ldo + col] = (half_t)val;
        } else if (MODE == 1) {
          int c = row / NCF, f = row % NCF;
          size_t idx = ((((size_t)blockIdx.z * C_EEG + c) * JDIM) + 1 + f) * MDIM + col;
          outF[idx] = val + extra[idx];
        } else if (MODE == 2) {
          float v2 = val + bias[col];
          float g  = 0.5f * v2 * (1.f + erff(v2 * 0.70710678118f));
          outH[(size_t)row * ldo + col] = (half_t)g;
        } else {
          outF[(size_t)row * MDIM + col] += val + bias[col];
        }
      }
    }
  }
}

// ---------------------------------------------------------------------------
// Fused attention per (b, i, h):  S = (Q/sqrt(Dh)) K^T ;  O = S V  (no softmax,
// faithful to the source).  Epilogue: ss += O * wos[h*60+d].
// ---------------------------------------------------------------------------
__global__ __launch_bounds__(256) void attn_kernel(
    const half_t* __restrict__ qkv,   // [ntok, 1440] = [.., x*480 + h*60 + d]
    float* __restrict__ ss,           // [ntok, 480]
    const float* __restrict__ wos,    // [480]  Wo.sum(axis=1)
    float scale)
{
  __shared__ half_t sm[26624];        // 53248 bytes
  half_t* Qs = sm;                    // 128 x 64, ld 72
  half_t* Ks = sm + 9216;             // 128 x 64, ld 72
  half_t* Ss = sm;                    // 128 x 128, ld 136 (reuses Q/K space)
  half_t* Vs = sm + 18432;            // 128 x 64, ld 64

  const int h  = blockIdx.x & 7;
  const int bi = blockIdx.x >> 3;     // b*127 + i

  // stage 1: load Q,K,V tiles (pad j>=121, d>=60 with zeros); scale Q
  {
    int j  = threadIdx.x >> 1;
    int d0 = (threadIdx.x & 1) * 32;
    bool jv = (j < JDIM);
    const half_t* base = qkv + ((size_t)bi * JDIM + (jv ? j : 0)) * 1440 + h * DH;
#pragma unroll 8
    for (int e = 0; e < 32; ++e) {
      int d = d0 + e;
      bool ok = jv && (d < DH);
      Qs[j * 72 + d] = ok ? (half_t)((float)base[d] * scale) : (half_t)0.f;
      Ks[j * 72 + d] = ok ? base[480 + d] : (half_t)0.f;
      Vs[j * 64 + d] = ok ? base[960 + d] : (half_t)0.f;
    }
  }
  __syncthreads();

  // stage 2: S = Q K^T (wave wid owns 16 rows x 128 cols), K = 64
  const int wid  = threadIdx.x >> 5;
  const int lane = threadIdx.x & 31;
  const int rsel = (lane >> 4) * 8;
  const int cn   = lane & 15;

  v8f s[8] = {};
  {
    v16h a0 = ld_frag_row(Qs, 72, 16 * wid, 0);
    v16h a1 = ld_frag_row(Qs, 72, 16 * wid, 32);
#pragma unroll
    for (int c = 0; c < 8; ++c) {
      v16h b0 = ld_frag_row(Ks, 72, 16 * c, 0);
      v16h b1 = ld_frag_row(Ks, 72, 16 * c, 32);
      s[c] = wmma_f32(a0, b0, s[c]);
      s[c] = wmma_f32(a1, b1, s[c]);
    }
  }
  __syncthreads();                    // done reading Q/K; S may overwrite them
#pragma unroll
  for (int c = 0; c < 8; ++c)
#pragma unroll
    for (int e = 0; e < 8; ++e)
      Ss[(size_t)(16 * wid + rsel + e) * 136 + 16 * c + cn] = (half_t)s[c][e];
  __syncthreads();

  // stage 3: O = S V  (16 rows x 64 cols per wave), K = 128
  v8f o[4] = {};
#pragma unroll
  for (int kk = 0; kk < 4; ++kk) {
    v16h a = ld_frag_row(Ss, 136, 16 * wid, 32 * kk);
#pragma unroll
    for (int n = 0; n < 4; ++n) {
      v16h b = ld_frag_col(Vs, 64, 16 * n, 32 * kk);
      o[n] = wmma_f32(a, b, o[n]);
    }
  }

  // epilogue: ss[b,i,j, h*60+d] += O * wos  (per-(b,i,h) columns are disjoint)
#pragma unroll
  for (int n = 0; n < 4; ++n) {
    int d = 16 * n + cn;
    if (d >= DH) continue;
    float wsc = wos[h * DH + d];
#pragma unroll
    for (int e = 0; e < 8; ++e) {
      int jr = 16 * wid + rsel + e;
      if (jr >= JDIM) continue;
      size_t idx = ((size_t)bi * JDIM + jr) * MDIM + h * DH + d;
      ss[idx] += o[n][e] * wsc;
    }
  }
}

// ---------------------------------------------------------------------------
// LayerNorm over last dim (480), f32 in -> f16 out.  One wave per token.
// ---------------------------------------------------------------------------
__global__ __launch_bounds__(256) void ln_kernel(
    const float* __restrict__ x, const float* __restrict__ g,
    const float* __restrict__ b, half_t* __restrict__ out, int ntok)
{
  int tok = blockIdx.x * 8 + (threadIdx.x >> 5);
  if (tok >= ntok) return;
  int lane = threadIdx.x & 31;
  const float* row = x + (size_t)tok * MDIM;
  float s = 0.f, s2 = 0.f;
  for (int e = lane; e < MDIM; e += 32) { float v = row[e]; s += v; s2 += v * v; }
#pragma unroll
  for (int off = 16; off; off >>= 1) {
    s  += __shfl_xor(s, off, 32);
    s2 += __shfl_xor(s2, off, 32);
  }
  float mu  = s / (float)MDIM;
  float var = s2 / (float)MDIM - mu * mu;
  float inv = rsqrtf(var + 1e-5f);
  half_t* orow = out + (size_t)tok * MDIM;
  for (int e = lane; e < MDIM; e += 32)
    orow[e] = (half_t)(((row[e] - mu) * inv) * g[e] + b[e]);
}

// ---------------------------------------------------------------------------
// Depthwise valid conv1d (k=10) + ReLU.  perc = out-channels per input channel.
// ---------------------------------------------------------------------------
template <typename OutT>
__global__ __launch_bounds__(256) void dwconv_relu_kernel(
    const float* __restrict__ x, const float* __restrict__ w,
    const float* __restrict__ bias, OutT* __restrict__ y,
    int C, int Tin, int Tout, int perc, int total)
{
  int idx = blockIdx.x * 256 + threadIdx.x;
  if (idx >= total) return;
  int t    = idx % Tout;
  int rest = idx / Tout;
  int oc   = rest % (C * perc);
  int b    = rest / (C * perc);
  int ic   = oc / perc;
  const float* xs = x + ((size_t)b * C + ic) * Tin + t;
  const float* ws = w + (size_t)oc * 10;
  float a = bias[oc];
#pragma unroll
  for (int k = 0; k < 10; ++k) a += xs[k] * ws[k];
  y[idx] = (OutT)fmaxf(a, 0.f);
}

__global__ void f32_to_f16_kernel(const float* __restrict__ in,
                                  half_t* __restrict__ out, int n) {
  int i = blockIdx.x * 256 + threadIdx.x;
  if (i < n) out[i] = (half_t)in[i];
}

// wos[kk*480+n] = sum_m Wo[kk][n, m]
__global__ void wosum_kernel(const float* __restrict__ Wo, float* __restrict__ wos) {
  int i = blockIdx.x * 256 + threadIdx.x;
  if (i >= 3 * MDIM) return;
  int kk = i / MDIM, n = i % MDIM;
  const float* p = Wo + (size_t)kk * MDIM * MDIM + (size_t)n * MDIM;
  float a = 0.f;
  for (int m = 0; m < MDIM; ++m) a += p[m];
  wos[i] = a;
}

// ss[b,c,0,:] = rtm_cls[b,c,0,:] + rtm_bias[b,c,0,:]
__global__ void cls_kernel(const float* __restrict__ cls,
                           const float* __restrict__ bias,
                           float* __restrict__ ssb) {
  int i = blockIdx.x * 256 + threadIdx.x;
  if (i >= BQ * C_EEG * MDIM) return;
  int n = i % MDIM; int bc = i / MDIM;
  size_t idx = ((size_t)bc * JDIM) * MDIM + n;
  ssb[idx] = cls[i] + bias[idx];
}

// t1[b,s,md] = relu(sum_c ss[b,md,s,c] * dw1[c] + db1)
__global__ __launch_bounds__(256) void dec1_kernel(
    const float* __restrict__ ss, const float* __restrict__ dw1,
    const float* __restrict__ db1, float* __restrict__ t1, int total)
{
  int idx = blockIdx.x * 8 + (threadIdx.x >> 5);
  if (idx >= total) return;
  int lane = threadIdx.x & 31;
  int md = idx % C_EEG; int rem = idx / C_EEG;
  int s = rem % JDIM;   int b = rem / JDIM;
  const float* row = ss + (((size_t)b * C_EEG + md) * JDIM + s) * MDIM;
  float a = 0.f;
  for (int c = lane; c < MDIM; c += 32) a += row[c] * dw1[c];
#pragma unroll
  for (int off = 16; off; off >>= 1) a += __shfl_xor(a, off, 32);
  if (lane == 0) t1[idx] = fmaxf(a + db1[0], 0.f);
}

// decoder tail + classifier, one block per batch sample
__global__ __launch_bounds__(128) void dec2_kernel(
    const float* __restrict__ t1,
    const float* __restrict__ dw2, const float* __restrict__ db2,
    const float* __restrict__ dw3, const float* __restrict__ db3,
    const float* __restrict__ fw,  const float* __restrict__ fb,
    float* __restrict__ out)
{
  __shared__ float t2s[C_EEG * 2];
  __shared__ float t3s[63 * 2];
  int b = blockIdx.x, tid = threadIdx.x;
  if (tid < C_EEG) {
    for (int n = 0; n < 2; ++n) {
      float a = db2[n];
      for (int s = 0; s < JDIM; ++s)
        a += t1[((size_t)b * JDIM + s) * C_EEG + tid] * dw2[n * JDIM + s];
      t2s[tid * 2 + n] = fmaxf(a, 0.f);
    }
  }
  __syncthreads();
  if (tid < 63) {
    for (int n = 0; n < 2; ++n) {
      float a = db3[tid];
      for (int md = 0; md < C_EEG; ++md)
        a += dw3[tid * C_EEG + md] * t2s[md * 2 + n];
      t3s[tid * 2 + n] = fmaxf(a, 0.f);
    }
  }
  __syncthreads();
  if (tid < 4) {
    float a = fb[tid];
    for (int q = 0; q < 126; ++q) a += t3s[q] * fw[tid * 126 + q];
    out[b * 4 + tid] = a;
  }
}

// ---------------------------------------------------------------------------
extern "C" void kernel_launch(void* const* d_in, const int* in_sizes, int n_in,
                              void* d_out, int out_size, void* d_ws, size_t ws_size,
                              hipStream_t stream) {
  (void)in_sizes; (void)n_in; (void)out_size; (void)ws_size;
  const float* x        = (const float*)d_in[0];
  const float* cw1      = (const float*)d_in[1];
  const float* cb1      = (const float*)d_in[2];
  const float* cw2      = (const float*)d_in[3];
  const float* cb2      = (const float*)d_in[4];
  const float* cw3      = (const float*)d_in[5];
  const float* cb3      = (const float*)d_in[6];
  const float* rtm_w    = (const float*)d_in[7];
  const float* rtm_bias = (const float*)d_in[8];
  const float* rtm_cls  = (const float*)d_in[9];
  const float* Wqkv     = (const float*)d_in[10];
  const float* Wo       = (const float*)d_in[11];
  const float* ln1g     = (const float*)d_in[12];
  const float* ln1b     = (const float*)d_in[13];
  const float* ln2g     = (const float*)d_in[14];
  const float* ln2b     = (const float*)d_in[15];
  const float* mW1      = (const float*)d_in[16];
  const float* mb1      = (const float*)d_in[17];
  const float* mW2      = (const float*)d_in[18];
  const float* mb2      = (const float*)d_in[19];
  const float* dw1      = (const float*)d_in[20];
  const float* db1      = (const float*)d_in[21];
  const float* dw2      = (const float*)d_in[22];
  const float* db2      = (const float*)d_in[23];
  const float* dw3      = (const float*)d_in[24];
  const float* db3      = (const float*)d_in[25];
  const float* fw       = (const float*)d_in[26];
  const float* fb       = (const float*)d_in[27];
  float* out = (float*)d_out;

  char* ws = (char*)d_ws;
  size_t off = 0;
  auto alloc = [&](size_t bytes) -> char* {
    size_t o = (off + 255) & ~(size_t)255;
    off = o + bytes;
    return ws + o;
  };

  const int ntok = NTOK;
  float*  ssb  = (float*)alloc((size_t)ntok * MDIM * 4);        // 118 MB
  half_t* hh   = (half_t*)alloc((size_t)ntok * MDIM * 2);       //  59 MB
  char*   Ureg = alloc((size_t)ntok * 1920 * 2);                // 236 MB union
  half_t* y3h  = (half_t*)Ureg;   // conv3 out  [4*15240, 480] f16
  half_t* qkvb = (half_t*)Ureg;   // qkv        [ntok, 1440]   f16
  half_t* tmlp = (half_t*)Ureg;   // mlp hidden [ntok, 1920]   f16
  float*  y1   = (float*)alloc((size_t)BQ * C_EEG * 498 * 4);
  float*  y2   = (float*)alloc((size_t)BQ * C_EEG * 489 * 4);
  half_t* wq16 = (half_t*)alloc((size_t)3 * 1440 * 480 * 2);
  half_t* w116 = (half_t*)alloc((size_t)3 * 1920 * 480 * 2);
  half_t* w216 = (half_t*)alloc((size_t)3 * 480 * 1920 * 2);
  half_t* rtw16= (half_t*)alloc((size_t)BQ * 480 * 480 * 2);
  float*  wosb = (float*)alloc((size_t)3 * MDIM * 4);
  float*  t1b  = (float*)alloc((size_t)ntok * 4);

  // ---- ODCM convs ----
  { int tot = BQ * C_EEG * 498;
    dwconv_relu_kernel<float><<<(tot + 255) / 256, 256, 0, stream>>>(
        x, cw1, cb1, y1, C_EEG, 507, 498, 1, tot); }
  { int tot = BQ * C_EEG * 489;
    dwconv_relu_kernel<float><<<(tot + 255) / 256, 256, 0, stream>>>(
        y1, cw2, cb2, y2, C_EEG, 498, 489, 1, tot); }
  { int tot = BQ * C_EEG * NCF * 480;
    dwconv_relu_kernel<half_t><<<(tot + 255) / 256, 256, 0, stream>>>(
        y2, cw3, cb3, y3h, C_EEG, 489, 480, NCF, tot); }

  // ---- weight conversions to f16 ----
  { int n = BQ * 480 * 480;  f32_to_f16_kernel<<<(n + 255) / 256, 256, 0, stream>>>(rtm_w, rtw16, n); }
  { int n = 3 * 1440 * 480;  f32_to_f16_kernel<<<(n + 255) / 256, 256, 0, stream>>>(Wqkv, wq16, n); }
  { int n = 3 * 1920 * 480;  f32_to_f16_kernel<<<(n + 255) / 256, 256, 0, stream>>>(mW1, w116, n); }
  { int n = 3 * 480 * 1920;  f32_to_f16_kernel<<<(n + 255) / 256, 256, 0, stream>>>(mW2, w216, n); }
  wosum_kernel<<<(3 * MDIM + 255) / 256, 256, 0, stream>>>(Wo, wosb);

  // ---- RTM embedding: cls row + GEMM scatter into ss ----
  cls_kernel<<<(BQ * C_EEG * MDIM + 255) / 256, 256, 0, stream>>>(rtm_cls, rtm_bias, ssb);
  {
    dim3 g((15240 + 127) / 128, (480 + 127) / 128, BQ);
    gemm_wmma_kernel<1><<<g, 256, 0, stream>>>(
        y3h, (long long)15240 * 480, 480,
        rtw16, (long long)480 * 480, 480,
        15240, 480, 480, ssb, nullptr, 0, nullptr, rtm_bias);
  }

  // ---- transformer blocks ----
  const int MB = (ntok + 127) / 128;   // 481
  for (int kk = 0; kk < 3; ++kk) {
    ln_kernel<<<(ntok + 7) / 8, 256, 0, stream>>>(ssb, ln1g + kk * MDIM,
                                                  ln1b + kk * MDIM, hh, ntok);
    { dim3 g(MB, (1440 + 127) / 128, 1);
      gemm_wmma_kernel<0><<<g, 256, 0, stream>>>(
          hh, 0, 480, wq16 + (size_t)kk * 1440 * 480, 0, 480,
          ntok, 1440, 480, nullptr, qkvb, 1440, nullptr, nullptr); }
    attn_kernel<<<BQ * C_EEG * HEADS, 256, 0, stream>>>(
        qkvb, ssb, wosb + kk * MDIM, 0.12909944487f /* 1/sqrt(60) */);
    ln_kernel<<<(ntok + 7) / 8, 256, 0, stream>>>(ssb, ln2g + kk * MDIM,
                                                  ln2b + kk * MDIM, hh, ntok);
    { dim3 g(MB, (1920 + 127) / 128, 1);
      gemm_wmma_kernel<2><<<g, 256, 0, stream>>>(
          hh, 0, 480, w116 + (size_t)kk * 1920 * 480, 0, 480,
          ntok, 1920, 480, nullptr, tmlp, 1920, mb1 + kk * 1920, nullptr); }
    { dim3 g(MB, (480 + 127) / 128, 1);
      gemm_wmma_kernel<3><<<g, 256, 0, stream>>>(
          tmlp, 0, 1920, w216 + (size_t)kk * 480 * 1920, 0, 1920,
          ntok, 480, 1920, ssb, nullptr, 0, mb2 + kk * 480, nullptr); }
  }

  // ---- CNN decoder + classifier ----
  dec1_kernel<<<(ntok + 7) / 8, 256, 0, stream>>>(ssb, dw1, db1, t1b, ntok);
  dec2_kernel<<<BQ, 128, 0, stream>>>(t1b, dw2, db2, dw3, db3, fw, fb, out);
}
